// BeliefPropagationCV_33414845562977
// MI455X (gfx1250) — matrix operand order repaired and checked by reading, not compiled, not run on hardware
//
#include <hip/hip_runtime.h>
#include <cmath>

typedef __attribute__((ext_vector_type(16))) _Float16 v16h;
typedef __attribute__((ext_vector_type(8)))  _Float16 v8h;
typedef __attribute__((ext_vector_type(2)))  _Float16 v2h;
typedef __attribute__((ext_vector_type(8)))  float    v8f;
typedef __attribute__((ext_vector_type(2)))  float    v2f;

constexpr int BATCH   = 128;
constexpr int OUT_DIM = 1024;
constexpr int IN_DIM  = 2048;

// ---------------------------------------------------------------------------
// Kernel 1: x[b,i] (f32) -> alog[b,i] = (f16)log|x|, aneg[b,i] = (f16)(x<0)
// ---------------------------------------------------------------------------
__global__ void bp_prep(const float* __restrict__ x,
                        _Float16* __restrict__ alog,
                        _Float16* __restrict__ aneg) {
  int i = blockIdx.x * blockDim.x + threadIdx.x;
  float v = x[i];
  float m = fmaxf(fabsf(v), 1e-12f);   // log(0) guard; exp(sum) -> ~0 -> out ~0
  alog[i] = (_Float16)__logf(m);
  aneg[i] = (_Float16)(v < 0.0f ? 1.0f : 0.0f);
}

// ---------------------------------------------------------------------------
// Kernel 2: two WMMA GEMMs [128 x 2048] x maskT[2048 x 1024] sharing the B
// operand, staged through LDS, then the atanh epilogue.
// Block = 256 threads = 8 wave32; block -> one 16-wide N tile (64 blocks),
// wave w -> M tile w (batch rows 16w..16w+15). K loop: 64 steps of 32.
// ---------------------------------------------------------------------------
__global__ __launch_bounds__(256) void bp_gemm(
    const _Float16* __restrict__ alog,
    const _Float16* __restrict__ aneg,
    const float*    __restrict__ mask,   // [OUT_DIM x IN_DIM], row-major
    float*          __restrict__ out) {  // [BATCH x OUT_DIM], row-major

  // B fragment staging buffer: per-lane contiguous 16 halves (32B), so each
  // wave lane reads its v16h as two ds_load_b128.
  __shared__ alignas(32) _Float16 bsh[32 * 16];

  const int t     = threadIdx.x;
  const int lane  = t & 31;
  const int wave  = t >> 5;          // M tile index 0..7
  const int ntile = blockIdx.x;      // N tile index 0..63
  const int ncol  = lane & 15;       // N within tile (B/C/D lane mapping)
  const int khalf = lane >> 4;       // lane-group half

  v8f acc_log = {};
  v8f acc_neg = {};

  // --- staging role of this thread: element pair (sl, 2*sj) of the B frag ---
  // B frag half-index sl*16 + 2j  <->  B[k][n] with n = sl&15, k = 8*(sl>>4)+2j+8*(j>>2)
  // and B[k][n] = maskT[k][n] = mask[n][k].
  const int sl    = t >> 3;                          // 0..31 (frag "lane")
  const int sj    = t & 7;                           // 0..7  (frag VGPR j)
  const int srow  = ntile * 16 + (sl & 15);          // mask row (check node)
  const int skoff = (sl >> 4) * 8 + sj * 2 + (sj >> 2) * 8; // k offset within tile
  const float* mrow = mask + (size_t)srow * IN_DIM + skoff;

  // --- A rows for this wave's lane: A frag lane holds row M = lane&15,
  //     K = {kbase+8h .. +7} U {kbase+16+8h .. +7} (two contiguous 16B runs)
  const int arow = wave * 16 + ncol;
  const _Float16* alog_row = alog + (size_t)arow * IN_DIM;
  const _Float16* aneg_row = aneg + (size_t)arow * IN_DIM;

  for (int ks = 0; ks < IN_DIM / 32; ++ks) {
    const int kbase = ks * 32;

    __syncthreads();                       // B reads of prev step done
    // stage + f32->f16 convert the 16x32 mask tile into B-frag order
    v2f mv = *(const v2f*)(mrow + kbase);
    v2h hv;
    hv.x = (_Float16)mv.x;
    hv.y = (_Float16)mv.y;
    *(v2h*)&bsh[sl * 16 + sj * 2] = hv;
    if (ks + 1 < IN_DIM / 32)
      __builtin_prefetch(mrow + kbase + 32, 0, 0);   // global_prefetch_b8
    __syncthreads();                       // B tile visible to all waves

    // A fragments (global, served from L2 after first block touches them)
    const int k0 = kbase + khalf * 8;
    v16h af, anf;
    *(v8h*)&af         = *(const v8h*)(alog_row + k0);
    *(((v8h*)&af) + 1) = *(const v8h*)(alog_row + k0 + 16);
    *(v8h*)&anf        = *(const v8h*)(aneg_row + k0);
    *(((v8h*)&anf) + 1)= *(const v8h*)(aneg_row + k0 + 16);

    // B fragment from LDS (same data broadcast to all 8 waves)
    v16h bf;
    *(v8h*)&bf         = *(const v8h*)&bsh[lane * 16];
    *(((v8h*)&bf) + 1) = *(const v8h*)&bsh[lane * 16 + 8];

    // D = A*B + C : log-magnitude sum and negative-count (exact in f32)
    acc_log = __builtin_amdgcn_wmma_f32_16x16x32_f16(
        false, af,  false, bf, (short)0, acc_log, false, false);
    acc_neg = __builtin_amdgcn_wmma_f32_16x16x32_f16(
        false, anf, false, bf, (short)0, acc_neg, false, false);
  }

  // --- epilogue: C/D layout VGPR r: lanes0-15 M=r, lanes16-31 M=8+r; N=lane&15
  const int n_out = ntile * 16 + ncol;
  #pragma unroll
  for (int r = 0; r < 8; ++r) {
    const int m  = wave * 16 + khalf * 8 + r;     // batch row
    float ls  = acc_log[r];
    float nc  = acc_neg[r];                        // exact integer count
    float mag = __expf(ls);
    int parity = ((int)(nc + 0.5f)) & 1;
    float p = parity ? -mag : mag;
    const float lim = 1.0f - 1e-7f;
    p = fminf(fmaxf(p, -lim), lim);
    out[(size_t)m * OUT_DIM + n_out] = __logf((1.0f + p) / (1.0f - p));
  }
}

// ---------------------------------------------------------------------------
extern "C" void kernel_launch(void* const* d_in, const int* in_sizes, int n_in,
                              void* d_out, int out_size, void* d_ws, size_t ws_size,
                              hipStream_t stream) {
  (void)in_sizes; (void)n_in; (void)out_size; (void)ws_size;

  const float* x    = (const float*)d_in[0];   // [BATCH, IN_DIM] f32
  const float* mask = (const float*)d_in[1];   // [OUT_DIM, IN_DIM] f32
  float* out        = (float*)d_out;           // [BATCH, OUT_DIM] f32

  // workspace: alog | aneg, each BATCH*IN_DIM f16  (1 MB total)
  _Float16* alog = (_Float16*)d_ws;
  _Float16* aneg = alog + (size_t)BATCH * IN_DIM;

  bp_prep<<<(BATCH * IN_DIM) / 256, 256, 0, stream>>>(x, alog, aneg);
  bp_gemm<<<OUT_DIM / 16, 256, 0, stream>>>(alog, aneg, mask, out);
}
